// ModifiedModelA_58007828300484
// MI455X (gfx1250) — matrix operand order repaired and checked by reading, not compiled, not run on hardware
//
#include <hip/hip_runtime.h>
#include <hip/hip_bf16.h>
#include <math.h>

typedef _Float16 f16_t;
typedef __attribute__((ext_vector_type(16))) _Float16 v16h;
typedef __attribute__((ext_vector_type(8)))  float    v8f;

#define BS   16
#define NN   350
#define KNBR 8
#define NPAIR (NN*NN)                 // 122500
#define NEDGE (2*NN*KNBR + NN)        // 5950
#define TILES_PER ((NPAIR + 15)/16)   // 7657
#define EITER 4                       // tiles per wave in edge kernel
#define C4K   1152                    // 128*9
#define C4M   (BS*256)                // 4096 output pixels total

// ---------------- conv + relu + 2x2 maxpool (fused, fp32: conv1-3) ----------------
__global__ void conv_relu_pool_k(const float* __restrict__ in, const float* __restrict__ w,
                                 const float* __restrict__ bias, float* __restrict__ out,
                                 int Cin, int Cout, int Hin, int Win) {
  int Ho = Hin >> 1, Wo = Win >> 1;
  int idx = blockIdx.x * blockDim.x + threadIdx.x;
  int total = BS * Cout * Ho * Wo;
  if (idx >= total) return;
  int ox = idx % Wo; int t = idx / Wo;
  int oy = t % Ho;   t /= Ho;
  int co = t % Cout; int b = t / Cout;
  const float* wb = w + (size_t)co * Cin * 9;
  float m = 0.f;  // all candidates relu'd (>=0)
  #pragma unroll
  for (int py = 0; py < 2; ++py)
  for (int px = 0; px < 2; ++px) {
    int y = 2*oy + py, x = 2*ox + px;
    float acc = bias[co];
    for (int ci = 0; ci < Cin; ++ci) {
      const float* ip = in + ((size_t)(b*Cin + ci) * Hin) * Win;
      const float* wp = wb + ci * 9;
      #pragma unroll
      for (int ky = 0; ky < 3; ++ky) {
        int yy = y + ky - 1;
        if ((unsigned)yy >= (unsigned)Hin) continue;
        #pragma unroll
        for (int kx = 0; kx < 3; ++kx) {
          int xx = x + kx - 1;
          if ((unsigned)xx >= (unsigned)Win) continue;
          acc += ip[yy*Win + xx] * wp[ky*3 + kx];
        }
      }
    }
    float r = acc > 0.f ? acc : 0.f;
    m = r > m ? r : m;
  }
  out[idx] = m;
}

// ---------------- im2col for conv4 (f32 -> f16, k = ci*9+ky*3+kx) ----------------
__global__ void im2col_k(const float* __restrict__ x3, f16_t* __restrict__ xc) {
  int idx = blockIdx.x * blockDim.x + threadIdx.x;
  int total = BS * 256 * C4K;
  if (idx >= total) return;
  int k = idx % C4K; int t = idx / C4K;
  int pix = t % 256; int b = t / 256;
  int ci = k / 9, r = k % 9, ky = r / 3, kx = r % 3;
  int y = pix >> 4, x = pix & 15;
  int yy = y + ky - 1, xx = x + kx - 1;
  float v = 0.f;
  if ((unsigned)yy < 16u && (unsigned)xx < 16u)
    v = x3[((size_t)(b*128 + ci) * 16 + yy) * 16 + xx];
  xc[idx] = (f16_t)v;
}

// ---------------- conv4 as implicit-GEMM WMMA ----------------
// A: [4096][1152] f16 (im2col), W: [256][1152] f16, outG: [4096][256] relu(A.W^T + b)
// Weight slab for this n-tile staged LDS via async global->LDS copies (ASYNCcnt).
__global__ void __launch_bounds__(256)
conv4_wmma_k(const f16_t* __restrict__ A, const f16_t* __restrict__ W,
             const float* __restrict__ bias, float* __restrict__ outG) {
  __shared__ __align__(16) f16_t sB[16 * C4K];   // 36 KB
  int tid = threadIdx.x;
  int ntile = blockIdx.x & 15;
  int mchunk = blockIdx.x >> 4;

  const f16_t* wsrc = W + (size_t)ntile * 16 * C4K;
  #pragma unroll
  for (int q = 0; q < 9; ++q) {                  // 2304 x 16B chunks / 256 threads
    int halfoff = (tid + 256*q) * 8;
    unsigned lds = (unsigned)(unsigned long long)(const void*)&sB[halfoff];
    unsigned long long ga = (unsigned long long)(const void*)(wsrc + halfoff);
    asm volatile("global_load_async_to_lds_b128 %0, %1, off" :: "v"(lds), "v"(ga) : "memory");
  }
  asm volatile("s_wait_asynccnt 0x0" ::: "memory");
  __syncthreads();

  int lane = tid & 31;
  int wave = tid >> 5;
  int mtile = mchunk * 8 + wave;
  int hiHalf = (lane >= 16) ? 1 : 0;
  int ncol = lane & 15;
  int m = mtile*16 + ncol;                       // A row (both lane halves, different K)
  const f16_t* arow = A + (size_t)m * C4K;
  __builtin_prefetch(arow, 0, 0);
  int kbA = hiHalf * 8, kbB = hiHalf * 16;
  v8f acc = {};
  for (int ks = 0; ks < C4K/32; ++ks) {
    v16h a, bv;
    #pragma unroll
    for (int e = 0; e < 16; ++e) {
      int Kk = ks*32 + (e & 7) + ((e & 8) ? 16 : 0) + kbA;
      a[e]  = arow[Kk];
      bv[e] = sB[ncol*C4K + Kk];
    }
    acc = __builtin_amdgcn_wmma_f32_16x16x32_f16(false, a, false, bv,
                                                 (short)0, acc, false, false);
  }
  int nglob = ntile*16 + ncol;
  float bb = bias[nglob];
  #pragma unroll
  for (int r = 0; r < 8; ++r) {
    int mrow = mtile*16 + r + 8*hiHalf;
    float v = acc[r] + bb;
    outG[(size_t)mrow*256 + nglob] = v > 0.f ? v : 0.f;
  }
}

// ---------------- GAP over [M][N] gemm output: g[b][c] = mean_p outG[b*256+p][c] ----
__global__ void gap2_k(const float* __restrict__ outG, float* __restrict__ g) {
  int idx = blockIdx.x * blockDim.x + threadIdx.x;
  if (idx >= BS * 256) return;
  int b = idx / 256, c = idx % 256;
  float s = 0.f;
  for (int p = 0; p < 256; ++p) s += outG[((size_t)b*256 + p)*256 + c];
  g[idx] = s * (1.f / 256.f);
}

// ---------------- generic linear (y = x @ W^T + b, optional relu) ----------------
__global__ void linear_k(const float* __restrict__ in, const float* __restrict__ w,
                         const float* __restrict__ bias, float* __restrict__ out,
                         int Bn, int IN, int OUT, int doRelu) {
  int idx = blockIdx.x * blockDim.x + threadIdx.x;
  if (idx >= Bn * OUT) return;
  int b = idx / OUT, o = idx % OUT;
  float acc = bias ? bias[o] : 0.f;
  const float* ip = in + (size_t)b * IN;
  const float* wp = w  + (size_t)o * IN;
  for (int k = 0; k < IN; ++k) acc += ip[k] * wp[k];
  out[idx] = doRelu ? fmaxf(acc, 0.f) : acc;
}

// ---------------- KNN (8 nearest, excluding nearest/self) ----------------
__global__ void knn_k(const float* __restrict__ coords, int* __restrict__ knn) {
  int idx = blockIdx.x * blockDim.x + threadIdx.x;
  if (idx >= BS * NN) return;
  int b = idx / NN, i = idx % NN;
  const float* cb = coords + (size_t)b * NN * 2;
  float xi = cb[i*2], yi = cb[i*2+1];
  float bd[KNBR+1]; int bj[KNBR+1];
  #pragma unroll
  for (int q = 0; q <= KNBR; ++q) { bd[q] = 3.4e38f; bj[q] = 0; }
  for (int j = 0; j < NN; ++j) {
    float dx = xi - cb[j*2], dy = yi - cb[j*2+1];
    float d2 = dx*dx + dy*dy;
    if (d2 < bd[KNBR]) {
      int q = KNBR;
      while (q > 0 && d2 < bd[q-1]) { bd[q] = bd[q-1]; bj[q] = bj[q-1]; --q; }
      bd[q] = d2; bj[q] = j;
    }
  }
  #pragma unroll
  for (int q = 0; q < KNBR; ++q) knn[(size_t)idx * KNBR + q] = bj[q+1];
}

__global__ void fill_k(float* __restrict__ p, float v, int n) {
  int idx = blockIdx.x * blockDim.x + threadIdx.x;
  if (idx < n) p[idx] = v;
}

__global__ void indeg_k(const int* __restrict__ knn, float* __restrict__ deg) {
  int idx = blockIdx.x * blockDim.x + threadIdx.x;
  if (idx >= BS * NN) return;
  int b = idx / NN;
  #pragma unroll
  for (int q = 0; q < KNBR; ++q)
    atomicAdd(&deg[(size_t)b * NN + knn[(size_t)idx * KNBR + q]], 1.0f);
}

__global__ void dinv_k(const float* __restrict__ deg, float* __restrict__ dinv, int n) {
  int idx = blockIdx.x * blockDim.x + threadIdx.x;
  if (idx < n) dinv[idx] = rsqrtf(fmaxf(deg[idx], 1e-12f));
}

// ---------------- normalized-adjacency scatter (segment_sum) ----------------
__global__ void gcn_scatter_k(const float* __restrict__ hw, const int* __restrict__ knn,
                              const float* __restrict__ dinv, float* __restrict__ agg, int C) {
  int idx = blockIdx.x * blockDim.x + threadIdx.x;
  if (idx >= BS * NEDGE) return;
  int b = idx / NEDGE, e = idx % NEDGE;
  int row, col;
  if (e < NN*KNBR) {
    row = e / KNBR;
    col = knn[((size_t)b*NN + row)*KNBR + (e % KNBR)];
  } else if (e < 2*NN*KNBR) {
    int e2 = e - NN*KNBR;
    int s = e2 / KNBR;
    row = knn[((size_t)b*NN + s)*KNBR + (e2 % KNBR)];
    col = s;
  } else {
    row = col = e - 2*NN*KNBR;
  }
  float nrm = dinv[b*NN + row] * dinv[b*NN + col];
  const float* src = hw  + ((size_t)b*NN + row) * C;
  float*       dst = agg + ((size_t)b*NN + col) * C;
  for (int c = 0; c < C; ++c) atomicAdd(&dst[c], nrm * src[c]);
}

__global__ void bias_relu_k(const float* __restrict__ agg, const float* __restrict__ bias,
                            float* __restrict__ h, int C, int n) {
  int idx = blockIdx.x * blockDim.x + threadIdx.x;
  if (idx >= n) return;
  float v = agg[idx] + bias[idx % C];
  h[idx] = v > 0.f ? v : 0.f;
}

__global__ void f2h_k(const float* __restrict__ src, f16_t* __restrict__ dst, int n) {
  int idx = blockIdx.x * blockDim.x + threadIdx.x;
  if (idx < n) dst[idx] = (f16_t)src[idx];
}

// ---------------- edge MLP: WMMA over all 350x350 pairs ----------------
// W1 fragments hoisted to VGPRs once per wave; EITER tiles of 16 pairs each.
__global__ void __launch_bounds__(256)
edge_wmma_k(const f16_t* __restrict__ h2h, const float* __restrict__ w1,
            const float* __restrict__ b1, const float* __restrict__ w2,
            const float* __restrict__ b2, float* __restrict__ adj) {
  __shared__ f16_t sW1[64*64];
  __shared__ float sB1[64];
  __shared__ float sW2[64];
  int tid = threadIdx.x;
  for (int t = tid; t < 64*64; t += 256) sW1[t] = (f16_t)w1[t];
  if (tid < 64) { sB1[tid] = b1[tid]; sW2[tid] = w2[tid]; }
  __syncthreads();

  int lane = tid & 31;
  int wave = tid >> 5;
  long group = (long)blockIdx.x * 8 + wave;
  const long totalTiles = (long)BS * TILES_PER;
  if (group * EITER >= totalTiles) return;       // wave-uniform

  int hiHalf = (lane >= 16) ? 1 : 0;
  int ncol = lane & 15;
  int kbA = hiHalf * 8;
  int kbB = hiHalf * 16;

  // hoist B (W1^T fragments) into registers: bre[nt*2+ks]
  v16h bre[8];
  #pragma unroll
  for (int nt = 0; nt < 4; ++nt) {
    #pragma unroll
    for (int ks = 0; ks < 2; ++ks) {
      #pragma unroll
      for (int e = 0; e < 16; ++e)
        bre[nt*2+ks][e] = sW1[(ncol + 16*nt)*64 + (e & 7) + ((e & 8) ? 16 : 0) + kbA + 32*ks];
    }
  }

  for (int it = 0; it < EITER; ++it) {
    long tg = group * EITER + it;
    if (tg >= totalTiles) break;                 // wave-uniform
    int bsamp = (int)(tg / TILES_PER);
    int t     = (int)(tg % TILES_PER);

    int p  = t*16 + ncol;
    int pc = p < NPAIR ? p : NPAIR - 1;          // clamp; EXEC stays all-ones
    int ii = pc / NN, jj = pc % NN;
    const f16_t* hri = h2h + ((size_t)bsamp*NN + ii) * 32;
    const f16_t* hrj = h2h + ((size_t)bsamp*NN + jj) * 32;

    v16h a0, a1;
    #pragma unroll
    for (int e = 0; e < 16; ++e) {
      int Kk = (e & 7) + ((e & 8) ? 16 : 0) + kbA;
      a0[e] = hri[Kk];                           // features 0..31  = h[i]
      a1[e] = hrj[Kk];                           // features 32..63 = h[j]
    }

    v8f acc[4] = {};
    #pragma unroll
    for (int nt = 0; nt < 4; ++nt) {
      acc[nt] = __builtin_amdgcn_wmma_f32_16x16x32_f16(false, a0, false, bre[nt*2+0],
                                                       (short)0, acc[nt], false, false);
      acc[nt] = __builtin_amdgcn_wmma_f32_16x16x32_f16(false, a1, false, bre[nt*2+1],
                                                       (short)0, acc[nt], false, false);
    }

    // layer 2: z[m] = sum_n relu(D[m][n]+b1[n]) * w2[n]
    float part[8];
    #pragma unroll
    for (int r = 0; r < 8; ++r) part[r] = 0.f;
    #pragma unroll
    for (int nt = 0; nt < 4; ++nt) {
      int n = ncol + 16*nt;
      float bb = sB1[n], ww = sW2[n];
      #pragma unroll
      for (int r = 0; r < 8; ++r) {
        float hv = acc[nt][r] + bb;
        part[r] += (hv > 0.f ? hv : 0.f) * ww;
      }
    }
    // reduce across each 16-lane half (xor 1,2,4,8 never crosses lane bit 4)
    #pragma unroll
    for (int off = 1; off < 16; off <<= 1) {
      #pragma unroll
      for (int r = 0; r < 8; ++r) part[r] += __shfl_xor(part[r], off, 32);
    }

    if ((lane & 15) < 8) {
      int r    = lane & 7;
      int mrow = r + 8*hiHalf;
      int pp   = t*16 + mrow;
      if (pp < NPAIR) {
        int i2 = pp / NN, j2 = pp % NN;
        float z  = part[r] + b2[0];
        float pv = 1.f / (1.f + expf(-z));
        float* ab = adj + (size_t)bsamp * NPAIR;
        if (i2 < j2)       { ab[i2*NN + j2] = pv; ab[j2*NN + i2] = pv; }
        else if (i2 == j2) { ab[i2*NN + j2] = 0.f; }
      }
    }
  }
}

// =====================================================================
extern "C" void kernel_launch(void* const* d_in, const int* in_sizes, int n_in,
                              void* d_out, int out_size, void* d_ws, size_t ws_size,
                              hipStream_t stream) {
  const float* images = (const float*)d_in[0];
  const float* c1w = (const float*)d_in[2];  const float* c1b = (const float*)d_in[3];
  const float* c2w = (const float*)d_in[4];  const float* c2b = (const float*)d_in[5];
  const float* c3w = (const float*)d_in[6];  const float* c3b = (const float*)d_in[7];
  const float* c4w = (const float*)d_in[8];  const float* c4b = (const float*)d_in[9];
  const float* np1w = (const float*)d_in[10]; const float* np1b = (const float*)d_in[11];
  const float* np2w = (const float*)d_in[12]; const float* np2b = (const float*)d_in[13];
  const float* cp1w = (const float*)d_in[14]; const float* cp1b = (const float*)d_in[15];
  const float* cp2w = (const float*)d_in[16]; const float* cp2b = (const float*)d_in[17];
  const float* g1w = (const float*)d_in[18]; const float* g1b = (const float*)d_in[19];
  const float* g2w = (const float*)d_in[20]; const float* g2b = (const float*)d_in[21];
  const float* e1w = (const float*)d_in[22]; const float* e1b = (const float*)d_in[23];
  const float* e2w = (const float*)d_in[24]; const float* e2b = (const float*)d_in[25];

  float* out    = (float*)d_out;
  float* coords = out;                       // 16*700
  float* adj    = out + BS*NN*2;             // 16*350*350
  float* count  = adj + (size_t)BS*NPAIR;    // 16

  // workspace carve-up (floats; all region sizes multiples of 4 -> 16B aligned)
  float* ws = (float*)d_ws;  size_t o = 0;
  float* X1  = ws + o;  o += (size_t)BS*32*64*64;
  float* X2  = ws + o;  o += (size_t)BS*64*32*32;
  float* X3  = ws + o;  o += (size_t)BS*128*16*16;
  float* X4G = ws + o;  o += (size_t)C4M*256;            // [4096][256] gemm out
  f16_t* X3C = (f16_t*)(ws + o); o += (size_t)C4M*C4K/2; // im2col f16
  f16_t* W4H = (f16_t*)(ws + o); o += (size_t)256*C4K/2; // conv4 weights f16
  float* G   = ws + o;  o += BS*256;
  float* T512 = ws + o; o += BS*512;
  float* T256 = ws + o; o += BS*256;
  float* DEG  = ws + o; o += BS*NN;
  float* DINV = ws + o; o += BS*NN;
  float* HW   = ws + o; o += (size_t)BS*NN*64;
  float* AGG  = ws + o; o += (size_t)BS*NN*64;
  float* H1   = ws + o; o += (size_t)BS*NN*64;
  float* H2   = ws + o; o += (size_t)BS*NN*32;
  f16_t* H2H  = (f16_t*)(ws + o); o += (size_t)BS*NN*32/2;
  int*   KNN  = (int*)(ws + o);   o += (size_t)BS*NN*KNBR;

  const int T = 256;
  #define GRD(n) dim3(((n) + T - 1) / T)

  // CNN backbone: conv1-3 direct fp32, conv4 as WMMA implicit GEMM
  conv_relu_pool_k<<<GRD(BS*32*64*64), T, 0, stream>>>(images, c1w, c1b, X1, 1, 32, 128, 128);
  conv_relu_pool_k<<<GRD(BS*64*32*32), T, 0, stream>>>(X1, c2w, c2b, X2, 32, 64, 64, 64);
  conv_relu_pool_k<<<GRD(BS*128*16*16), T, 0, stream>>>(X2, c3w, c3b, X3, 64, 128, 32, 32);
  f2h_k   <<<GRD(256*C4K), T, 0, stream>>>(c4w, W4H, 256*C4K);
  im2col_k<<<GRD(C4M*C4K), T, 0, stream>>>(X3, X3C);
  conv4_wmma_k<<<dim3(16*32), 256, 0, stream>>>(X3C, W4H, c4b, X4G);  // 16 ntiles x 32 mchunks
  gap2_k  <<<GRD(BS*256), T, 0, stream>>>(X4G, G);

  // heads
  linear_k<<<GRD(BS*512), T, 0, stream>>>(G, np1w, np1b, T512, BS, 256, 512, 1);
  linear_k<<<GRD(BS*700), T, 0, stream>>>(T512, np2w, np2b, coords, BS, 512, 700, 0);
  linear_k<<<GRD(BS*256), T, 0, stream>>>(G, cp1w, cp1b, T256, BS, 256, 256, 1);
  linear_k<<<GRD(BS*1),   T, 0, stream>>>(T256, cp2w, cp2b, count, BS, 256, 1, 0);

  // KNN graph + symmetric normalization
  knn_k  <<<GRD(BS*NN), T, 0, stream>>>(coords, KNN);
  fill_k <<<GRD(BS*NN), T, 0, stream>>>(DEG, 9.0f, BS*NN);   // self + K src appearances
  indeg_k<<<GRD(BS*NN), T, 0, stream>>>(KNN, DEG);
  dinv_k <<<GRD(BS*NN), T, 0, stream>>>(DEG, DINV, BS*NN);

  // GCN layer 1 (2 -> 64)
  linear_k<<<GRD(BS*NN*64), T, 0, stream>>>(coords, g1w, nullptr, HW, BS*NN, 2, 64, 0);
  fill_k  <<<GRD(BS*NN*64), T, 0, stream>>>(AGG, 0.f, BS*NN*64);
  gcn_scatter_k<<<GRD(BS*NEDGE), T, 0, stream>>>(HW, KNN, DINV, AGG, 64);
  bias_relu_k  <<<GRD(BS*NN*64), T, 0, stream>>>(AGG, g1b, H1, 64, BS*NN*64);

  // GCN layer 2 (64 -> 32)
  linear_k<<<GRD(BS*NN*32), T, 0, stream>>>(H1, g2w, nullptr, HW, BS*NN, 64, 32, 0);
  fill_k  <<<GRD(BS*NN*32), T, 0, stream>>>(AGG, 0.f, BS*NN*32);
  gcn_scatter_k<<<GRD(BS*NEDGE), T, 0, stream>>>(HW, KNN, DINV, AGG, 32);
  bias_relu_k  <<<GRD(BS*NN*32), T, 0, stream>>>(AGG, g2b, H2, 32, BS*NN*32);
  f2h_k        <<<GRD(BS*NN*32), T, 0, stream>>>(H2, H2H, BS*NN*32);

  // edge MLP over all pairs — WMMA, B-resident, EITER tiles/wave
  long totalTiles = (long)BS * TILES_PER;                    // 122512
  long groups     = (totalTiles + EITER - 1) / EITER;        // 30628
  int  edgeBlocks = (int)((groups + 7) / 8);
  edge_wmma_k<<<edgeBlocks, 256, 0, stream>>>(H2H, e1w, e1b, e2w, e2b, adj);
  #undef GRD
}
